// HeteroSAGEEncoder_16492674417215
// MI455X (gfx1250) — compile-verified, matrix operand order checked by reading, not run on hardware
//
#include <hip/hip_runtime.h>
#include <hip/hip_bf16.h>
#include <stdint.h>
#include <stddef.h>

typedef __bf16 bf16_t;
typedef __attribute__((ext_vector_type(16))) __bf16 v16bf;
typedef __attribute__((ext_vector_type(8)))  float  v8f;

union Frag { v16bf v; float4 f4[2]; };

#define N_OUT 256

// ---------------- utility kernels ----------------

__global__ void zero_f32_kernel(float* p, size_t n) {
    size_t i = (size_t)blockIdx.x * blockDim.x + threadIdx.x;
    size_t stride = (size_t)gridDim.x * blockDim.x;
    for (; i < n; i += stride) p[i] = 0.0f;
}

__global__ void count_deg_kernel(const int* __restrict__ idx, int nE, float* deg) {
    int t = blockIdx.x * blockDim.x + threadIdx.x;
    if (t < nE) atomicAdd(&deg[idx[t]], 1.0f);
}

__global__ void inv_deg_kernel(float* deg, int n) {
    int t = blockIdx.x * blockDim.x + threadIdx.x;
    if (t < n) deg[t] = 1.0f / fmaxf(deg[t], 1.0f);
}

// scatter-add f32 features, D = 128 floats/row = 32 x float4 chunks.
// One wave per edge; lane = 16B chunk. e = t>>5, c = t&31 (no division).
__global__ void scatter_f32_kernel(const float* __restrict__ X,
                                   const int* __restrict__ srcI,
                                   const int* __restrict__ dstI,
                                   int nE, float* agg) {
    int t = blockIdx.x * blockDim.x + threadIdx.x;
    int e = t >> 5;
    if (e >= nE) return;
    int c = (t & 31) << 2;                    // float offset
    int s = srcI[e], d = dstI[e];
    float4 v = *(const float4*)(X + (size_t)s * 128 + c);
    float* o = agg + (size_t)d * 128 + c;
    atomicAdd(o + 0, v.x); atomicAdd(o + 1, v.y);
    atomicAdd(o + 2, v.z); atomicAdd(o + 3, v.w);
}

// scatter-add bf16 features, D = 256 bf16/row = 32 x 16B chunks.
__global__ void scatter_bf16_kernel(const bf16_t* __restrict__ X,
                                    const int* __restrict__ srcI,
                                    const int* __restrict__ dstI,
                                    int nE, float* agg) {
    int t = blockIdx.x * blockDim.x + threadIdx.x;
    int e = t >> 5;
    if (e >= nE) return;
    int c = (t & 31) << 3;                    // bf16 offset
    int s = srcI[e], d = dstI[e];
    union { uint4 u; bf16_t b[8]; } v;
    v.u = *(const uint4*)(X + (size_t)s * 256 + c);
    float* o = agg + (size_t)d * 256 + c;
#pragma unroll
    for (int i = 0; i < 8; ++i) atomicAdd(o + i, (float)v.b[i]);
}

// Transpose + bf16-convert one K-segment of a weight matrix into Bt[N][Ktot].
__global__ void prep_bt_kernel(bf16_t* bt, const float* __restrict__ W0,
                               const float* __restrict__ W1,
                               int Dseg, int Ktot, int kOff) {
    int t = blockIdx.x * blockDim.x + threadIdx.x;
    if (t >= Dseg * N_OUT) return;
    int k = t >> 8;
    int n = t & 255;
    float v = W0[k * N_OUT + n];
    if (W1) v += W1[k * N_OUT + n];
    bt[(size_t)n * Ktot + kOff + k] = (bf16_t)v;
}

__global__ void combine_bias_kernel(float* o, const float* a, const float* b) {
    int t = threadIdx.x;
    o[t] = a[t] + b[t];
}

// ---------------- fused SAGE GEMM (bf16 WMMA, f32 accumulate) ----------------
// Out[M,256] = sum_s rowscale_s(m) * Aseg_s[m,:] @ Bt_seg_s + bias, opt ReLU.
// A segments: row-major, width (1<<dShift) each, f32 or bf16 (bfMask bit/seg).
// Bt: bf16 [256][Ktot] pre-transposed. Block 256 thr (8 waves), tile 128x64,
// BK=64 (8 WMMAs per stage). B tile staged with async load-to-LDS (ASYNCcnt).
__launch_bounds__(256)
__global__ void gemm_sage_wmma(const void* A0, const void* A1, const void* A2,
                               const float* s0, const float* s1, const float* s2,
                               int bfMask, int dShift, int M,
                               const bf16_t* __restrict__ Bt, int Ktot,
                               const float* __restrict__ bias,
                               void* Out, int outBf16, int relu) {
    __shared__ __align__(16) bf16_t lA[128][72];   // 144B rows: 16B-aligned chunks
    __shared__ __align__(16) bf16_t lB[64][72];

    const void*  As[3]  = {A0, A1, A2};
    const float* scs[3] = {s0, s1, s2};
    const int dMask = (1 << dShift) - 1;

    const int tid    = threadIdx.x;
    const int blockM = blockIdx.x * 128;
    const int nBase  = blockIdx.y * 64;

    const int lane  = tid & 31;
    const int wv    = tid >> 5;
    const int warpM = wv >> 1;          // 0..3
    const int warpN = wv & 1;           // 0..1
    const int lo    = lane & 15;
    const int hi8   = (lane >> 4) << 3; // 0 or 8

    v8f c00 = {0, 0, 0, 0, 0, 0, 0, 0};
    v8f c01 = c00, c10 = c00, c11 = c00;

    for (int k0 = 0; k0 < Ktot; k0 += 64) {
        __syncthreads();
        // ---- stage B tile: 64 n-rows x 64 k via async global->LDS (16B/lane) ----
#pragma unroll
        for (int q0 = 0; q0 < 2; ++q0) {
            int q  = tid + q0 * 256;        // 0..511
            int n  = q >> 3;                // 0..63
            int kc = (q & 7) << 3;          // 0..56
            unsigned ldsOff = (unsigned)(uintptr_t)&lB[n][kc];
            unsigned gOff   = (unsigned)(((size_t)(nBase + n) * Ktot + k0 + kc) * 2);
            asm volatile("global_load_async_to_lds_b128 %0, %1, %2"
                         :: "v"(ldsOff), "v"(gOff), "s"(Bt) : "memory");
        }
        // ---- stage A tile: 128 rows x 64 k, mean-scale + f32->bf16 fused ----
#pragma unroll
        for (int q0 = 0; q0 < 4; ++q0) {
            int q   = tid + q0 * 256;       // 0..1023
            int row = q >> 3;               // 0..127
            int kc  = (q & 7) << 3;         // 0..56
            int kG  = k0 + kc;
            int s   = kG >> dShift;         // segment (power-of-2 width)
            int kSeg = kG & dMask;
            int rG  = blockM + row;
            bf16_t* dst = &lA[row][kc];
            if (rG < M) {
                const float* scp = scs[s];
                float sc = scp ? scp[rG] : 1.0f;
                if (bfMask & (1 << s)) {
                    union { uint4 u; bf16_t b[8]; } v;
                    v.u = *(const uint4*)((const bf16_t*)As[s] + (size_t)rG * (size_t)(dMask + 1) + kSeg);
#pragma unroll
                    for (int i = 0; i < 8; ++i) dst[i] = (bf16_t)((float)v.b[i] * sc);
                } else {
                    const float* p = (const float*)As[s] + (size_t)rG * (size_t)(dMask + 1) + kSeg;
                    float4 x = *(const float4*)p;
                    float4 y = *(const float4*)(p + 4);
                    dst[0] = (bf16_t)(x.x * sc); dst[1] = (bf16_t)(x.y * sc);
                    dst[2] = (bf16_t)(x.z * sc); dst[3] = (bf16_t)(x.w * sc);
                    dst[4] = (bf16_t)(y.x * sc); dst[5] = (bf16_t)(y.y * sc);
                    dst[6] = (bf16_t)(y.z * sc); dst[7] = (bf16_t)(y.w * sc);
                }
            } else {
#pragma unroll
                for (int i = 0; i < 8; ++i) dst[i] = (bf16_t)0.0f;
            }
        }
#if __has_builtin(__builtin_amdgcn_s_wait_asynccnt)
        __builtin_amdgcn_s_wait_asynccnt(0);
#else
        asm volatile("s_wait_asynccnt 0x0" ::: "memory");
#endif
        __syncthreads();

        // ---- 8 WMMAs per stage; ISA 16-bit frag layout (K 0-7/16-23 vs 8-15/24-31) ----
#pragma unroll
        for (int kk = 0; kk < 64; kk += 32) {
            Frag a0, a1, b0, b1;
            int ar = warpM * 32 + lo;
            a0.f4[0] = *(const float4*)&lA[ar][kk + hi8];
            a0.f4[1] = *(const float4*)&lA[ar][kk + 16 + hi8];
            a1.f4[0] = *(const float4*)&lA[ar + 16][kk + hi8];
            a1.f4[1] = *(const float4*)&lA[ar + 16][kk + 16 + hi8];
            int bc = warpN * 32 + lo;
            b0.f4[0] = *(const float4*)&lB[bc][kk + hi8];
            b0.f4[1] = *(const float4*)&lB[bc][kk + 16 + hi8];
            b1.f4[0] = *(const float4*)&lB[bc + 16][kk + hi8];
            b1.f4[1] = *(const float4*)&lB[bc + 16][kk + 16 + hi8];

            c00 = __builtin_amdgcn_wmma_f32_16x16x32_bf16(false, a0.v, false, b0.v, (short)0, c00, false, false);
            c01 = __builtin_amdgcn_wmma_f32_16x16x32_bf16(false, a0.v, false, b1.v, (short)0, c01, false, false);
            c10 = __builtin_amdgcn_wmma_f32_16x16x32_bf16(false, a1.v, false, b0.v, (short)0, c10, false, false);
            c11 = __builtin_amdgcn_wmma_f32_16x16x32_bf16(false, a1.v, false, b1.v, (short)0, c11, false, false);
        }
    }

    // ---- epilogue: bias (+ReLU); C layout: N = lane&15, M = vgpr + 8*(lane>>4) ----
#pragma unroll
    for (int i = 0; i < 2; ++i) {
#pragma unroll
        for (int j = 0; j < 2; ++j) {
            v8f c = (i == 0) ? (j == 0 ? c00 : c01) : (j == 0 ? c10 : c11);
            int nG = nBase + warpN * 32 + j * 16 + lo;
            float bv = bias[nG];
#pragma unroll
            for (int r = 0; r < 8; ++r) {
                int mG = blockM + warpM * 32 + i * 16 + hi8 + r;
                if (mG < M) {
                    float v = c[r] + bv;
                    if (relu) v = fmaxf(v, 0.0f);
                    if (outBf16)
                        ((bf16_t*)Out)[(size_t)mG * N_OUT + nG] = (bf16_t)v;
                    else
                        ((float*)Out)[(size_t)mG * N_OUT + nG] = v;
                }
            }
        }
    }
}

// ---------------- host orchestration ----------------

extern "C" void kernel_launch(void* const* d_in, const int* in_sizes, int n_in,
                              void* d_out, int out_size, void* d_ws, size_t ws_size,
                              hipStream_t stream) {
    (void)n_in; (void)out_size; (void)ws_size;
    const float* x_user   = (const float*)d_in[0];
    const float* x_item   = (const float*)d_in[1];
    const int*   src_rev  = (const int*)d_in[2];
    const int*   dst_rev  = (const int*)d_in[3];
    const int*   src_ab   = (const int*)d_in[4];
    const int*   dst_ab   = (const int*)d_in[5];
    const float* Wl1_rev  = (const float*)d_in[6];
    const float* Wr1_rev  = (const float*)d_in[7];
    const float* b1_rev   = (const float*)d_in[8];
    const float* Wl1_rrev = (const float*)d_in[9];
    const float* Wr1_rrev = (const float*)d_in[10];
    const float* b1_rrev  = (const float*)d_in[11];
    const float* Wl1_ab   = (const float*)d_in[12];
    const float* Wr1_ab   = (const float*)d_in[13];
    const float* b1_ab    = (const float*)d_in[14];
    const float* Wl2_rev  = (const float*)d_in[15];
    const float* Wr2_rev  = (const float*)d_in[16];
    const float* b2_rev   = (const float*)d_in[17];
    const float* Wl2_rrev = (const float*)d_in[18];
    const float* Wr2_rrev = (const float*)d_in[19];
    const float* b2_rrev  = (const float*)d_in[20];
    const float* Wl2_ab   = (const float*)d_in[21];
    const float* Wr2_ab   = (const float*)d_in[22];
    const float* b2_ab    = (const float*)d_in[23];

    const int D1 = 128;
    const int NU = in_sizes[0] / D1;
    const int NI = in_sizes[1] / D1;
    const int ER = in_sizes[2];
    const int EA = in_sizes[4];

    // ---- carve workspace (~412 MB) ----
    char* w = (char*)d_ws;
    size_t off = 0;
    auto carve = [&](size_t bytes) -> void* {
        void* p = w + off;
        off = (off + bytes + 255) & ~(size_t)255;
        return p;
    };
    float*  degA   = (float*)carve((size_t)NI * 4);
    float*  degB   = (float*)carve((size_t)NI * 4);
    float*  degC   = (float*)carve((size_t)NU * 4);
    float*  agg0   = (float*)carve((size_t)NI * 256 * 4);
    float*  agg1   = (float*)carve((size_t)NI * 256 * 4);
    float*  agg2   = (float*)carve((size_t)NU * 256 * 4);
    bf16_t* hItem  = (bf16_t*)carve((size_t)NI * 256 * 2);
    bf16_t* hUser  = (bf16_t*)carve((size_t)NU * 256 * 2);
    bf16_t* bt1i   = (bf16_t*)carve((size_t)256 * 384 * 2);
    bf16_t* bt1u   = (bf16_t*)carve((size_t)256 * 256 * 2);
    bf16_t* bt2i   = (bf16_t*)carve((size_t)256 * 768 * 2);
    bf16_t* bt2u   = (bf16_t*)carve((size_t)256 * 512 * 2);
    float*  bias1i = (float*)carve(256 * 4);
    float*  bias2i = (float*)carve(256 * 4);

    dim3 B(256);
    auto blks = [](long long n) { return dim3((unsigned)((n + 255) / 256)); };

    // ---- degrees (edge-structure only; shared by both layers) ----
    zero_f32_kernel<<<dim3(1024), B, 0, stream>>>(degA, (size_t)NI);
    zero_f32_kernel<<<dim3(1024), B, 0, stream>>>(degB, (size_t)NI);
    zero_f32_kernel<<<dim3(1024), B, 0, stream>>>(degC, (size_t)NU);
    count_deg_kernel<<<blks(ER), B, 0, stream>>>(dst_rev, ER, degA);
    count_deg_kernel<<<blks(EA), B, 0, stream>>>(dst_ab, EA, degB);
    count_deg_kernel<<<blks(ER), B, 0, stream>>>(src_rev, ER, degC);
    inv_deg_kernel<<<blks(NI), B, 0, stream>>>(degA, NI);
    inv_deg_kernel<<<blks(NI), B, 0, stream>>>(degB, NI);
    inv_deg_kernel<<<blks(NU), B, 0, stream>>>(degC, NU);

    // ---- weight prep: transpose to [N][Ktot] bf16, root weights pre-summed ----
    prep_bt_kernel<<<blks(128 * 256), B, 0, stream>>>(bt1i, Wl1_rev, nullptr, 128, 384, 0);
    prep_bt_kernel<<<blks(128 * 256), B, 0, stream>>>(bt1i, Wl1_ab,  nullptr, 128, 384, 128);
    prep_bt_kernel<<<blks(128 * 256), B, 0, stream>>>(bt1i, Wr1_rev, Wr1_ab,  128, 384, 256);
    prep_bt_kernel<<<blks(128 * 256), B, 0, stream>>>(bt1u, Wl1_rrev, nullptr, 128, 256, 0);
    prep_bt_kernel<<<blks(128 * 256), B, 0, stream>>>(bt1u, Wr1_rrev, nullptr, 128, 256, 128);
    prep_bt_kernel<<<blks(256 * 256), B, 0, stream>>>(bt2i, Wl2_rev, nullptr, 256, 768, 0);
    prep_bt_kernel<<<blks(256 * 256), B, 0, stream>>>(bt2i, Wl2_ab,  nullptr, 256, 768, 256);
    prep_bt_kernel<<<blks(256 * 256), B, 0, stream>>>(bt2i, Wr2_rev, Wr2_ab,  256, 768, 512);
    prep_bt_kernel<<<blks(256 * 256), B, 0, stream>>>(bt2u, Wl2_rrev, nullptr, 256, 512, 0);
    prep_bt_kernel<<<blks(256 * 256), B, 0, stream>>>(bt2u, Wr2_rrev, nullptr, 256, 512, 256);
    combine_bias_kernel<<<dim3(1), B, 0, stream>>>(bias1i, b1_rev, b1_ab);
    combine_bias_kernel<<<dim3(1), B, 0, stream>>>(bias2i, b2_rev, b2_ab);

    // ---- layer 1: aggregate (f32 gathers, D=128; one wave per edge) ----
    zero_f32_kernel<<<dim3(4096), B, 0, stream>>>(agg0, (size_t)NI * 128);
    zero_f32_kernel<<<dim3(4096), B, 0, stream>>>(agg1, (size_t)NI * 128);
    zero_f32_kernel<<<dim3(4096), B, 0, stream>>>(agg2, (size_t)NU * 128);
    scatter_f32_kernel<<<blks((long long)ER * 32), B, 0, stream>>>(x_user, src_rev, dst_rev, ER, agg0);
    scatter_f32_kernel<<<blks((long long)EA * 32), B, 0, stream>>>(x_item, src_ab, dst_ab, EA, agg1);
    scatter_f32_kernel<<<blks((long long)ER * 32), B, 0, stream>>>(x_item, dst_rev, src_rev, ER, agg2);

    // ---- layer 1: fused mean-scale + GEMM + bias + ReLU -> bf16 hidden ----
    gemm_sage_wmma<<<dim3((NI + 127) / 128, 4), B, 0, stream>>>(
        agg0, agg1, x_item, degA, degB, nullptr,
        /*bfMask*/0, /*dShift*/7, NI, bt1i, 384, bias1i, hItem, /*outBf16*/1, /*relu*/1);
    gemm_sage_wmma<<<dim3((NU + 127) / 128, 4), B, 0, stream>>>(
        agg2, x_user, nullptr, degC, nullptr, nullptr,
        0, 7, NU, bt1u, 256, b1_rrev, hUser, 1, 1);

    // ---- layer 2: aggregate (bf16 gathers, D=256; one wave per edge) ----
    zero_f32_kernel<<<dim3(4096), B, 0, stream>>>(agg0, (size_t)NI * 256);
    zero_f32_kernel<<<dim3(4096), B, 0, stream>>>(agg1, (size_t)NI * 256);
    zero_f32_kernel<<<dim3(4096), B, 0, stream>>>(agg2, (size_t)NU * 256);
    scatter_bf16_kernel<<<blks((long long)ER * 32), B, 0, stream>>>(hUser, src_rev, dst_rev, ER, agg0);
    scatter_bf16_kernel<<<blks((long long)EA * 32), B, 0, stream>>>(hItem, src_ab, dst_ab, EA, agg1);
    scatter_bf16_kernel<<<blks((long long)ER * 32), B, 0, stream>>>(hItem, dst_rev, src_rev, ER, agg2);

    // ---- layer 2: GEMM -> f32 outputs (o_user first, then o_item) ----
    float* oUser = (float*)d_out;
    float* oItem = (float*)d_out + (size_t)NU * 256;
    gemm_sage_wmma<<<dim3((NI + 127) / 128, 4), B, 0, stream>>>(
        agg0, agg1, hItem, degA, degB, nullptr,
        /*bfMask*/(1 << 2), /*dShift*/8, NI, bt2i, 768, bias2i, oItem, 0, 0);
    gemm_sage_wmma<<<dim3((NU + 127) / 128, 4), B, 0, stream>>>(
        agg2, hUser, nullptr, degC, nullptr, nullptr,
        (1 << 1), 8, NU, bt2u, 512, b2_rrev, oUser, 0, 0);
    (void)b2_ab;
}
